// MixtralDecoderLayer_60919816126696
// MI455X (gfx1250) — matrix operand order repaired and checked by reading, not compiled, not run on hardware
//
#include <hip/hip_runtime.h>
#include <hip/hip_bf16.h>

// ---------------------------------------------------------------------------
// Mixtral decoder layer for MI455X (gfx1250, wave32, WMMA).
//   T=2048 HID=1024 NH=8 NKV=2 HD=128 INTER=3584 NE=8 TOPK=2
// All GEMMs: bf16 WMMA (v_wmma_f32_16x16x32_bf16) with fp32 accumulation.
// Weights converted to bf16 once per call -> whole weight set (~180MB) is
// L2-resident (192MB). GEMM: 4-wave WGs, WG tile 32x256, wave tile 32x64,
// A tile staged through double-buffered LDS with async global->LDS copies
// (ASYNCcnt pipeline) so the next k-step's A is in flight during WMMA.
// ---------------------------------------------------------------------------

typedef unsigned short u16;
typedef __attribute__((ext_vector_type(16))) __bf16 v16bf;
typedef __attribute__((ext_vector_type(8)))  float  v8f;
typedef __attribute__((ext_vector_type(4)))  unsigned int v4u;
typedef int v4i_ __attribute__((vector_size(16)));  // matches builtin param type

constexpr int  kT    = 2048;
constexpr int  kHID  = 1024;
constexpr int  kNH   = 8;
constexpr int  kNKV  = 2;
constexpr int  kHD   = 128;
constexpr int  kQKVN = (kNH + 2 * kNKV) * kHD;  // 1536
constexpr int  kINTER= 3584;
constexpr int  kNE   = 8;
constexpr float kEPS   = 1e-5f;
constexpr float kTHETA = 10000.0f;
constexpr float kSCALE = 0.08838834764831845f;  // 1/sqrt(128)

#if defined(__AMDGCN__) && __has_builtin(__builtin_amdgcn_global_load_async_to_lds_b128)
#define HAS_ASYNC_LDS 1
#else
#define HAS_ASYNC_LDS 0
#endif

// ---------------- bf16 helpers (RNE) ----------------
__device__ __forceinline__ u16 f2bf(float f) {
  unsigned u = __builtin_bit_cast(unsigned, f);
  unsigned r = u + 0x7FFFu + ((u >> 16) & 1u);
  return (u16)(r >> 16);
}
__device__ __forceinline__ float bf2f(u16 h) {
  unsigned u = ((unsigned)h) << 16;
  return __builtin_bit_cast(float, u);
}

// 16x32 bf16 fragment: 8 VGPRs / lane, loaded as two 16B vectors.
union BFrag {
  v16bf v;
  v4u   q[2];
};

__device__ __forceinline__ v8f wmma_bf16(const BFrag& a, const BFrag& b, v8f c) {
  return __builtin_amdgcn_wmma_f32_16x16x32_bf16(
      /*neg_a=*/false, a.v, /*neg_b=*/false, b.v,
      /*c_mod=*/(short)0, c, /*reuse_a=*/false, /*reuse_b=*/false);
}

// 16B global -> LDS copy; async (GLOBAL_LOAD_ASYNC_TO_LDS_B128) when available.
__device__ __forceinline__ void copy16_g2l(const u16* g, u16* l) {
#if HAS_ASYNC_LDS
  __builtin_amdgcn_global_load_async_to_lds_b128(
      (__attribute__((address_space(1))) v4i_*)g,
      (__attribute__((address_space(3))) v4i_*)l,
      /*offset=*/0, /*cpol=*/0);
#else
  *reinterpret_cast<v4u*>(l) = *reinterpret_cast<const v4u*>(g);
#endif
}

__device__ __forceinline__ void wait_async0() {
#if HAS_ASYNC_LDS
#if __has_builtin(__builtin_amdgcn_s_wait_asynccnt)
  __builtin_amdgcn_s_wait_asynccnt(0);
#else
  asm volatile("s_wait_asynccnt 0x0" ::: "memory");
#endif
#endif
}

// ---------------- f32 -> bf16 weight conversion ----------------
__global__ __launch_bounds__(256) void cvt_f32_bf16(const float* __restrict__ in,
                                                    u16* __restrict__ out,
                                                    long long n) {
  long long i = (long long)blockIdx.x * blockDim.x + threadIdx.x;
  long long stride = (long long)gridDim.x * blockDim.x;
  for (; i < n; i += stride) out[i] = f2bf(in[i]);
}

__global__ __launch_bounds__(256) void zero_f32(float* __restrict__ p, long long n) {
  long long i = (long long)blockIdx.x * blockDim.x + threadIdx.x;
  long long stride = (long long)gridDim.x * blockDim.x;
  for (; i < n; i += stride) p[i] = 0.0f;
}

// ---------------- RMSNorm (one block per token row) ----------------
__global__ __launch_bounds__(256) void rmsnorm_bf16(const float* __restrict__ x,
                                                    const float* __restrict__ w,
                                                    u16* __restrict__ out) {
  const int t = blockIdx.x;
  const int tid = threadIdx.x;
  const float* row = x + (size_t)t * kHID;
  float ss = 0.0f;
  for (int k = tid; k < kHID; k += 256) { float v = row[k]; ss += v * v; }
  for (int off = 16; off > 0; off >>= 1) ss += __shfl_xor(ss, off);
  __shared__ float red[8];
  __shared__ float rinv_s;
  if ((tid & 31) == 0) red[tid >> 5] = ss;
  __syncthreads();
  if (tid == 0) {
    float tot = 0.0f;
    for (int i = 0; i < 8; ++i) tot += red[i];
    rinv_s = rsqrtf(tot / (float)kHID + kEPS);
  }
  __syncthreads();
  const float rinv = rinv_s;
  for (int k = tid; k < kHID; k += 256)
    out[(size_t)t * kHID + k] = f2bf(row[k] * rinv * w[k]);
}

// ---------------- WMMA GEMM: C[M,N] = A[M,K] * W[N,K]^T ----------------
// 4 waves per WG; WG tile 32(M) x 256(N); wave tile 32x64 (8 accumulators).
// A tile (32x32 bf16) double-buffered in LDS, filled with async g->LDS copies;
// B fragments stream straight from global (weights are L2-resident) with
// global_prefetch for the next k-step. Epilogues:
//   mode 0: Cf = v
//   mode 1: Cf = v + aux            (residual add; aux = hidden_states)
//   mode 2: Cb = bf16(silu(aux)*v)  (aux = gate GEMM result; SwiGLU)
//   mode 3: Cf += scale[m*NE] * v   (router-weighted MoE accumulation)
__global__ __launch_bounds__(128) void gemm_bf16_wmma(
    const u16* __restrict__ A, const u16* __restrict__ W,
    float* __restrict__ Cf, u16* __restrict__ Cb,
    const float* __restrict__ aux, const float* __restrict__ scale,
    int M, int N, int K, int mode) {
  __shared__ u16 sA[2][32 * 32];  // 2 x 2KB double buffer

  const int tid  = threadIdx.x;
  const int wave = tid >> 5;
  const int lane = tid & 31;
  const int l15  = lane & 15;
  const bool hi  = lane >= 16;
  const int m0 = blockIdx.y * 32;
  const int n0 = blockIdx.x * 256 + wave * 64;

  // Cooperative A-tile copy mapping: 128 threads x 16B = 32 rows x 64B.
  const int crow = tid >> 2;        // 0..31
  const int ccol = (tid & 3) * 8;   // element offset 0,8,16,24
  const u16* ag = A + (size_t)(m0 + crow) * K + ccol;

  const int nsteps = K / 32;

  // Preload k-step 0 into buffer 0.
  copy16_g2l(ag, &sA[0][crow * 32 + ccol]);

  v8f acc[2][4] = {};

  for (int s = 0; s < nsteps; ++s) {
    const int k0 = s * 32;
    const int buf = s & 1;

    wait_async0();     // this wave's async copies landed
    __syncthreads();   // all waves' copies landed; prev buffer fully consumed

    if (s + 1 < nsteps)
      copy16_g2l(ag + (k0 + 32), &sA[buf ^ 1][crow * 32 + ccol]);

    // A fragments for the two 16-row sub-tiles (ISA 16-bit A layout).
    BFrag a[2];
#pragma unroll
    for (int mt = 0; mt < 2; ++mt) {
      const u16* ar = &sA[buf][(mt * 16 + l15) * 32 + (hi ? 8 : 0)];
      a[mt].q[0] = *reinterpret_cast<const v4u*>(ar);
      a[mt].q[1] = *reinterpret_cast<const v4u*>(ar + 16);
    }

#pragma unroll
    for (int j = 0; j < 4; ++j) {
      const u16* wrow = W + (size_t)(n0 + j * 16 + l15) * K + k0 + (hi ? 8 : 0);
      __builtin_prefetch(wrow + 32);  // next k-step -> global_prefetch_b8
      BFrag b;
      b.q[0] = *reinterpret_cast<const v4u*>(wrow);
      b.q[1] = *reinterpret_cast<const v4u*>(wrow + 16);
      acc[0][j] = wmma_bf16(a[0], b, acc[0][j]);
      acc[1][j] = wmma_bf16(a[1], b, acc[1][j]);
    }
  }

  const int hio = hi ? 8 : 0;
#pragma unroll
  for (int mt = 0; mt < 2; ++mt) {
#pragma unroll
    for (int j = 0; j < 4; ++j) {
      const int col = n0 + j * 16 + l15;
#pragma unroll
      for (int r = 0; r < 8; ++r) {
        const int m = m0 + mt * 16 + r + hio;
        const size_t idx = (size_t)m * N + col;
        const float v = acc[mt][j][r];
        if (mode == 0) {
          Cf[idx] = v;
        } else if (mode == 1) {
          Cf[idx] = v + aux[idx];
        } else if (mode == 2) {
          const float g = aux[idx];
          const float h = (g / (1.0f + __expf(-g))) * v;  // silu(g)*u
          Cb[idx] = f2bf(h);
        } else {  // mode 3
          Cf[idx] += scale[(size_t)m * kNE] * v;
        }
      }
    }
  }
}

// ---------------- RoPE + QKV split / V transpose ----------------
__global__ __launch_bounds__(64) void rope_split(const int* __restrict__ pos,
                                                 const float* __restrict__ qkv,
                                                 u16* __restrict__ qb,
                                                 u16* __restrict__ kb,
                                                 u16* __restrict__ vtb) {
  const int t = blockIdx.x;
  const int i = threadIdx.x;  // 0..63 = rotary pair index
  const float p = (float)pos[t];
  const float freq = __powf(kTHETA, -(float)(2 * i) / (float)kHD);
  const float f = p * freq;
  const float c = __cosf(f), s = __sinf(f);
  const float* row = qkv + (size_t)t * kQKVN;

#pragma unroll
  for (int h = 0; h < kNH; ++h) {
    const float x1 = row[h * kHD + i];
    const float x2 = row[h * kHD + 64 + i];
    u16* q = qb + ((size_t)h * kT + t) * kHD;
    q[i]      = f2bf(x1 * c - x2 * s);
    q[64 + i] = f2bf(x2 * c + x1 * s);
  }
#pragma unroll
  for (int kv = 0; kv < kNKV; ++kv) {
    const float x1 = row[kNH * kHD + kv * kHD + i];
    const float x2 = row[kNH * kHD + kv * kHD + 64 + i];
    u16* k = kb + ((size_t)kv * kT + t) * kHD;
    k[i]      = f2bf(x1 * c - x2 * s);
    k[64 + i] = f2bf(x2 * c + x1 * s);
    const float v0 = row[(kNH + kNKV) * kHD + kv * kHD + i];
    const float v1 = row[(kNH + kNKV) * kHD + kv * kHD + 64 + i];
    vtb[((size_t)kv * kHD + i) * kT + t]      = f2bf(v0);
    vtb[((size_t)kv * kHD + i + 64) * kT + t] = f2bf(v1);
  }
}

// ---------------- Causal attention: one wave per (16-query tile, head) ----
// Dynamic LDS: 16*T f32 scores (128KB) + 16*T bf16 probs (64KB) = 192KB,
// legal under CDNA5's 320KB-per-workgroup LDS.
__global__ __launch_bounds__(32) void attn_wmma(const u16* __restrict__ Q,
                                                const u16* __restrict__ Kb,
                                                const u16* __restrict__ VT,
                                                u16* __restrict__ Aout) {
  extern __shared__ char smem[];
  float* sS = reinterpret_cast<float*>(smem);
  u16*   sP = reinterpret_cast<u16*>(smem + (size_t)16 * kT * sizeof(float));

  const int bx = blockIdx.x;        // query tile
  const int h  = blockIdx.y;        // head
  const int kv = h / (kNH / kNKV);  // grouped KV head
  const int m0 = bx * 16;
  const int L  = m0 + 16;           // causal key count
  const int lane = threadIdx.x & 31;
  const int l15  = lane & 15;
  const bool hi  = lane >= 16;
  const int hio  = hi ? 8 : 0;

  BFrag qf[4];
  const u16* qrow = Q + ((size_t)h * kT + m0 + l15) * kHD + (hi ? 8 : 0);
#pragma unroll
  for (int kc = 0; kc < 4; ++kc) {
    qf[kc].q[0] = *reinterpret_cast<const v4u*>(qrow + kc * 32);
    qf[kc].q[1] = *reinterpret_cast<const v4u*>(qrow + kc * 32 + 16);
  }

  // ---- S = Q K^T (scaled, causal-masked) into LDS ----
  const u16* Kh = Kb + (size_t)kv * kT * kHD;
  for (int kbidx = 0; kbidx <= bx; ++kbidx) {
    const int n0 = kbidx * 16;
    v8f acc = {};
#pragma unroll
    for (int kc = 0; kc < 4; ++kc) {
      const u16* kr = Kh + (size_t)(n0 + l15) * kHD + kc * 32 + (hi ? 8 : 0);
      BFrag b;
      b.q[0] = *reinterpret_cast<const v4u*>(kr);
      b.q[1] = *reinterpret_cast<const v4u*>(kr + 16);
      acc = wmma_bf16(qf[kc], b, acc);
    }
#pragma unroll
    for (int r = 0; r < 8; ++r) {
      const int m = r + hio;
      float val = acc[r] * kSCALE;
      if (n0 + l15 > m0 + m) val = -3.0e38f;  // causal mask
      sS[m * kT + n0 + l15] = val;
    }
  }
  __syncthreads();

  // ---- row softmax (2 lanes per row, interleaved columns) ----
  {
    const int row = l15;
    const int half = hi ? 1 : 0;
    float mx = -3.0e38f;
    for (int c = half; c < L; c += 2) mx = fmaxf(mx, sS[row * kT + c]);
    mx = fmaxf(mx, __shfl_xor(mx, 16));
    float sum = 0.0f;
    for (int c = half; c < L; c += 2) sum += __expf(sS[row * kT + c] - mx);
    sum += __shfl_xor(sum, 16);
    const float inv = 1.0f / sum;
    for (int c = half; c < L; c += 2)
      sP[row * kT + c] = f2bf(__expf(sS[row * kT + c] - mx) * inv);
  }
  const int Lpad = (L + 31) & ~31;
  if (Lpad > L) {
    for (int i = lane; i < 16 * 16; i += 32) {
      const int rr = i >> 4, cc = L + (i & 15);
      if (cc < Lpad) sP[rr * kT + cc] = 0;
    }
  }
  __syncthreads();

  // ---- O = P V  (V stored transposed: contiguous B fragments) ----
  const u16* Vh = VT + (size_t)kv * kHD * kT;
#pragma unroll
  for (int dt = 0; dt < 8; ++dt) {
    v8f acc = {};
    const u16* vrow = Vh + (size_t)(dt * 16 + l15) * kT + (hi ? 8 : 0);
    const u16* prow = sP + (size_t)l15 * kT + (hi ? 8 : 0);
    for (int n0 = 0; n0 < Lpad; n0 += 32) {
      BFrag a, b;
      a.q[0] = *reinterpret_cast<const v4u*>(prow + n0);
      a.q[1] = *reinterpret_cast<const v4u*>(prow + n0 + 16);
      b.q[0] = *reinterpret_cast<const v4u*>(vrow + n0);
      b.q[1] = *reinterpret_cast<const v4u*>(vrow + n0 + 16);
      acc = wmma_bf16(a, b, acc);
    }
#pragma unroll
    for (int r = 0; r < 8; ++r) {
      const int m = m0 + r + hio;
      Aout[(size_t)m * (kNH * kHD) + h * kHD + dt * 16 + l15] = f2bf(acc[r]);
    }
  }
}

// ---------------- Router: softmax + top-2 -> dense combine weights ----------
__global__ __launch_bounds__(32) void router_topk(const u16* __restrict__ x2,
                                                  const float* __restrict__ gate_w,
                                                  float* __restrict__ cw) {
  const int t = blockIdx.x;
  const int lane = threadIdx.x & 31;
  float logit[kNE];
#pragma unroll
  for (int e = 0; e < kNE; ++e) {
    float s = 0.0f;
    const u16* xr = x2 + (size_t)t * kHID;
    const float* gr = gate_w + (size_t)e * kHID;
    for (int k = lane; k < kHID; k += 32) s += bf2f(xr[k]) * gr[k];
    for (int off = 16; off > 0; off >>= 1) s += __shfl_xor(s, off);
    logit[e] = s;
  }
  if (lane == 0) {
    float mx = logit[0];
#pragma unroll
    for (int e = 1; e < kNE; ++e) mx = fmaxf(mx, logit[e]);
    float p[kNE], sum = 0.0f;
#pragma unroll
    for (int e = 0; e < kNE; ++e) { p[e] = __expf(logit[e] - mx); sum += p[e]; }
#pragma unroll
    for (int e = 0; e < kNE; ++e) p[e] /= sum;
    int i1 = 0;
#pragma unroll
    for (int e = 1; e < kNE; ++e) if (p[e] > p[i1]) i1 = e;
    int i2 = (i1 == 0) ? 1 : 0;
#pragma unroll
    for (int e = 0; e < kNE; ++e) if (e != i1 && p[e] > p[i2]) i2 = e;
    const float norm = 1.0f / (p[i1] + p[i2]);
#pragma unroll
    for (int e = 0; e < kNE; ++e)
      cw[(size_t)t * kNE + e] =
          (e == i1) ? p[i1] * norm : ((e == i2) ? p[i2] * norm : 0.0f);
  }
}

// ---------------------------------------------------------------------------
extern "C" void kernel_launch(void* const* d_in, const int* in_sizes, int n_in,
                              void* d_out, int out_size, void* d_ws, size_t ws_size,
                              hipStream_t stream) {
  (void)in_sizes; (void)n_in; (void)out_size; (void)ws_size;

  const int*   positions = (const int*)  d_in[0];
  const float* hidden    = (const float*)d_in[1];
  const float* qkv_w     = (const float*)d_in[2];
  const float* o_w       = (const float*)d_in[3];
  const float* gate_w    = (const float*)d_in[4];
  const float* w1        = (const float*)d_in[5];
  const float* w2        = (const float*)d_in[6];
  const float* w3        = (const float*)d_in[7];
  const float* ln1_w     = (const float*)d_in[8];
  const float* ln2_w     = (const float*)d_in[9];

  float* moe_out  = (float*)d_out;                       // [T][HID]
  float* residual = (float*)d_out + (size_t)kT * kHID;   // [T][HID]

  char* wsb = (char*)d_ws;
  size_t off = 0;
  auto alloc = [&](size_t bytes) -> void* {
    void* p = wsb + off;
    off += (bytes + 255) & ~(size_t)255;
    return p;
  };
  u16*   xn1    = (u16*)  alloc((size_t)kT * kHID * 2);
  float* qkvbuf = (float*)alloc((size_t)kT * kQKVN * 4);
  u16*   qb     = (u16*)  alloc((size_t)kNH * kT * kHD * 2);
  u16*   kbuf   = (u16*)  alloc((size_t)kNKV * kT * kHD * 2);
  u16*   vtb    = (u16*)  alloc((size_t)kNKV * kHD * kT * 2);
  u16*   attnb  = (u16*)  alloc((size_t)kT * kNH * kHD * 2);
  u16*   xn2    = (u16*)  alloc((size_t)kT * kHID * 2);
  float* gbuf   = (float*)alloc((size_t)kT * kINTER * 4);
  u16*   hbuf   = (u16*)  alloc((size_t)kT * kINTER * 2);
  float* cw     = (float*)alloc((size_t)kT * kNE * 4);
  u16*   qkv_wb = (u16*)  alloc((size_t)kQKVN * kHID * 2);
  u16*   o_wb   = (u16*)  alloc((size_t)kHID * (kNH * kHD) * 2);
  u16*   w1b    = (u16*)  alloc((size_t)kNE * kINTER * kHID * 2);
  u16*   w2b    = (u16*)  alloc((size_t)kNE * kHID * kINTER * 2);
  u16*   w3b    = (u16*)  alloc((size_t)kNE * kINTER * kHID * 2);

  // ---- weight conversion to bf16 (makes full weight set L2-resident) ----
  cvt_f32_bf16<<<2048, 256, 0, stream>>>(qkv_w, qkv_wb, (long long)kQKVN * kHID);
  cvt_f32_bf16<<<2048, 256, 0, stream>>>(o_w, o_wb, (long long)kHID * kNH * kHD);
  cvt_f32_bf16<<<4096, 256, 0, stream>>>(w1, w1b, (long long)kNE * kINTER * kHID);
  cvt_f32_bf16<<<4096, 256, 0, stream>>>(w2, w2b, (long long)kNE * kHID * kINTER);
  cvt_f32_bf16<<<4096, 256, 0, stream>>>(w3, w3b, (long long)kNE * kINTER * kHID);

  // ---- attention block ----
  rmsnorm_bf16<<<kT, 256, 0, stream>>>(hidden, ln1_w, xn1);

  gemm_bf16_wmma<<<dim3(kQKVN / 256, kT / 32), 128, 0, stream>>>(
      xn1, qkv_wb, qkvbuf, nullptr, nullptr, nullptr, kT, kQKVN, kHID, 0);

  rope_split<<<kT, 64, 0, stream>>>(positions, qkvbuf, qb, kbuf, vtb);

  const size_t attn_lds = (size_t)16 * kT * sizeof(float) + (size_t)16 * kT * sizeof(u16);
  attn_wmma<<<dim3(kT / 16, kNH), 32, attn_lds, stream>>>(qb, kbuf, vtb, attnb);

  // residual = hidden + attn @ o_w^T   (fused epilogue)
  gemm_bf16_wmma<<<dim3(kHID / 256, kT / 32), 128, 0, stream>>>(
      attnb, o_wb, residual, nullptr, hidden, nullptr, kT, kHID, kNH * kHD, 1);

  // ---- MoE block ----
  rmsnorm_bf16<<<kT, 256, 0, stream>>>(residual, ln2_w, xn2);
  router_topk<<<kT, 32, 0, stream>>>(xn2, gate_w, cw);
  zero_f32<<<2048, 256, 0, stream>>>(moe_out, (long long)kT * kHID);

  for (int e = 0; e < kNE; ++e) {
    const u16* w1e = w1b + (size_t)e * kINTER * kHID;
    const u16* w3e = w3b + (size_t)e * kINTER * kHID;
    const u16* w2e = w2b + (size_t)e * kHID * kINTER;
    // g = x @ w1^T
    gemm_bf16_wmma<<<dim3(kINTER / 256, kT / 32), 128, 0, stream>>>(
        xn2, w1e, gbuf, nullptr, nullptr, nullptr, kT, kINTER, kHID, 0);
    // h = silu(g) * (x @ w3^T), stored bf16
    gemm_bf16_wmma<<<dim3(kINTER / 256, kT / 32), 128, 0, stream>>>(
        xn2, w3e, nullptr, hbuf, gbuf, nullptr, kT, kINTER, kHID, 2);
    // moe_out += cw[:,e] * (h @ w2^T)
    gemm_bf16_wmma<<<dim3(kHID / 256, kT / 32), 128, 0, stream>>>(
        hbuf, w2e, moe_out, nullptr, nullptr, cw + e, kT, kHID, kINTER, 3);
  }
}
